// AttentionKernel_89455578841177
// MI455X (gfx1250) — matrix operand order repaired and checked by reading, compiled, not run on hardware
//
#include <hip/hip_runtime.h>
#include <hip/hip_bf16.h>

typedef __attribute__((ext_vector_type(16))) __bf16 v16bf;
typedef __attribute__((ext_vector_type(8)))  float  v8f;

#define BATCH 2
#define SEQ   2048
#define EDIM  1024
#define HEADS 16
#define DDIM  64
#define HD    (HEADS * DDIM)      // 1024
#define QKV_N (3 * HD)            // 3072

// exp2 scale: softmax(s/sqrt(64)) == exp2((s - m) * 0.125 * log2(e)) normalized
#define EXP2_SCALE 0.18033688011112042f   // 0.125 * log2(e)

__device__ __forceinline__ float fast_exp2(float x) {
#if __has_builtin(__builtin_amdgcn_exp2f)
    return __builtin_amdgcn_exp2f(x);
#else
    return exp2f(x);
#endif
}

__device__ __forceinline__ float fast_rcp(float x) {
#if __has_builtin(__builtin_amdgcn_rcpf)
    return __builtin_amdgcn_rcpf(x);
#else
    return 1.0f / x;
#endif
}

// scalar round-to-nearest-even f32 -> bf16 (bit pattern)
__device__ __forceinline__ unsigned short f32_bf16(float f) {
    unsigned int u = __float_as_uint(f);
    unsigned int r = (u + 0x7FFFu + ((u >> 16) & 1u)) >> 16;
    return (unsigned short)r;
}

__device__ __forceinline__ unsigned int pack_bf16(float lo, float hi) {
#if __has_builtin(__builtin_amdgcn_cvt_pk_bf16_f32)
    auto p = __builtin_amdgcn_cvt_pk_bf16_f32(lo, hi);
    union { decltype(p) v; unsigned int u; } c;
    c.v = p;
    return c.u;
#else
    return (unsigned int)f32_bf16(lo) | ((unsigned int)f32_bf16(hi) << 16);
#endif
}

union Frag { v16bf v; unsigned int u[8]; };

// constant all-ones bf16 B fragment (for row-sum via WMMA)
__device__ __forceinline__ v16bf ones_frag() {
    Frag f;
#pragma unroll
    for (int j = 0; j < 8; ++j) f.u[j] = 0x3F803F80u;   // two bf16 1.0
    return f.v;
}

// A-matrix fragment (16x32 bf16): half 0 -> K 0..7 & 16..23, half 1 -> K 8..15 & 24..31.
__device__ __forceinline__ v16bf load_a_frag(const unsigned int* row16, int half) {
    Frag f;
#pragma unroll
    for (int j = 0; j < 4; ++j) {
        f.u[j]     = row16[half * 4 + j];
        f.u[4 + j] = row16[8 + half * 4 + j];
    }
    return f.v;
}

// B-matrix fragment (32x16 bf16): half 0 -> K 0..7 & 16..23, half 1 -> K 8..15 & 24..31 of column idx.
__device__ __forceinline__ v16bf load_b_frag(const unsigned int* row16, int half) {
    Frag f;
#pragma unroll
    for (int j = 0; j < 4; ++j) {
        f.u[j]     = row16[half * 4 + j];
        f.u[4 + j] = row16[8 + half * 4 + j];
    }
    return f.v;
}

// 2x2 16-bit transpose: two dwords along-n at rows k,k+1 -> two dwords along-k at n,n+1
__device__ __forceinline__ void trans_pair_store(unsigned int* dst, int stride,
                                                 const uint4 a, const uint4 b) {
    const unsigned int av[4] = {a.x, a.y, a.z, a.w};
    const unsigned int bv[4] = {b.x, b.y, b.z, b.w};
#pragma unroll
    for (int e = 0; e < 4; ++e) {
        dst[(2 * e)     * stride] = (av[e] & 0xFFFFu) | (bv[e] << 16);
        dst[(2 * e + 1) * stride] = (av[e] >> 16)     | (bv[e] & 0xFFFF0000u);
    }
}

__device__ __forceinline__ void store_out(float* p, float v) { *p = v; }
__device__ __forceinline__ void store_out(unsigned short* p, float v) { *p = f32_bf16(v); }

// -------------------- one-time f32 -> bf16 conversion pass -------------------------
__global__ __launch_bounds__(256)
void cvt_f32_bf16(const float* __restrict__ src, unsigned short* __restrict__ dst, int n4) {
    int i = blockIdx.x * 256 + threadIdx.x;
    if (i < n4) {
        const float4 v = ((const float4*)src)[i];
        ((uint2*)dst)[i] = make_uint2(pack_bf16(v.x, v.y), pack_bf16(v.z, v.w));
    }
}

// -------------------- C = A[MxK] @ B[KxN], bf16 in, OutT out -----------------------
// block tile 128(M) x 128(N), K-step 64, 256 threads = 8 waves as 4(M) x 2(N),
// each wave computes 32x64 -> 16 WMMA per K-step.
template <int N, int K, typename OutT>
__global__ __launch_bounds__(256)
void gemm_bf16_wmma(const unsigned short* __restrict__ A,
                    const unsigned short* __restrict__ B, OutT* __restrict__ C) {
    __shared__ unsigned int ldsA[128 * 32];   // 128 rows x 64 bf16
    __shared__ unsigned int ldsB[128 * 32];   // 128 cols x 64 bf16 (B tile transposed)

    const int t    = threadIdx.x;
    const int wid  = t >> 5;
    const int lane = t & 31;
    const int half = lane >> 4;
    const int idx  = lane & 15;
    const int wm   = wid >> 1;                // 0..3
    const int wn   = wid & 1;                 // 0..1
    const int m0   = blockIdx.y * 128;
    const int n0   = blockIdx.x * 128;

    v8f acc[2][4];
#pragma unroll
    for (int i = 0; i < 2; ++i)
#pragma unroll
        for (int j = 0; j < 4; ++j) acc[i][j] = v8f{0.f,0.f,0.f,0.f,0.f,0.f,0.f,0.f};

    for (int k0 = 0; k0 < K; k0 += 64) {
        __syncthreads();
        // stage A tile 128x64 bf16: pure b128 copies, 4 per thread
#pragma unroll
        for (int i = 0; i < 4; ++i) {
            int l = t + i * 256;                  // 0..1023
            int r = l >> 3, c8 = l & 7;           // 8 x (8 bf16) per row
            const unsigned short* ap = A + (size_t)(m0 + r) * K + k0 + c8 * 8;
            if (k0 + 64 < K) __builtin_prefetch(ap + 64, 0, 1);
            *(uint4*)&ldsA[r * 32 + c8 * 4] = *(const uint4*)ap;
        }
        // stage B tile 64x128 bf16 transposed to [n][k]: row-pair 2x2 transposes
#pragma unroll
        for (int i = 0; i < 2; ++i) {
            int u = t + i * 256;                  // 0..511
            int kp = u >> 4, n8 = (u & 15) * 8;   // k-pair 0..31, n 0..120
            const unsigned short* bp = B + (size_t)(k0 + kp * 2) * N + n0 + n8;
            if (k0 + 64 < K) __builtin_prefetch(bp + (size_t)64 * N, 0, 1);
            const uint4 e0 = *(const uint4*)bp;
            const uint4 e1 = *(const uint4*)(bp + N);
            trans_pair_store(&ldsB[n8 * 32 + kp], 32, e0, e1);
        }
        __syncthreads();

#pragma unroll
        for (int ks = 0; ks < 2; ++ks) {
            v16bf a[2], b[4];
#pragma unroll
            for (int i = 0; i < 2; ++i)
                a[i] = load_a_frag(&ldsA[(wm * 32 + i * 16 + idx) * 32 + ks * 16], half);
#pragma unroll
            for (int j = 0; j < 4; ++j)
                b[j] = load_b_frag(&ldsB[(wn * 64 + j * 16 + idx) * 32 + ks * 16], half);
#pragma unroll
            for (int i = 0; i < 2; ++i)
#pragma unroll
                for (int j = 0; j < 4; ++j)
                    acc[i][j] = __builtin_amdgcn_wmma_f32_16x16x32_bf16(
                        false, a[i], false, b[j], (short)0, acc[i][j], false, false);
        }
    }

    // epilogue: constant N -> per-r offsets fold into store immediates
#pragma unroll
    for (int i = 0; i < 2; ++i)
#pragma unroll
        for (int j = 0; j < 4; ++j) {
            int row0 = m0 + wm * 32 + i * 16 + half * 8;
            int col  = n0 + wn * 64 + j * 16 + idx;
            OutT* cp = C + (size_t)row0 * N + col;
#pragma unroll
            for (int r = 0; r < 8; ++r) store_out(cp + r * N, acc[i][j][r]);
        }
}

// -------------------- flash attention: one (b, h, 128-query block) per workgroup ---
// 256 threads = 8 waves; each wave owns 16 query rows; bf16 qkv in, bf16 attn out.
// Row sums of P computed with WMMA against an all-ones fragment (no shuffle chain).
__global__ __launch_bounds__(256)
void attn_fwd(const unsigned short* __restrict__ qkv, unsigned short* __restrict__ out) {
    __shared__ unsigned int ldsQ[128 * 32];       // 128x64 bf16
    __shared__ unsigned int ldsK[64 * 32];        // 64 keys x 64 d (row-major)
    __shared__ unsigned int ldsVt[64 * 32];       // 64 d x 64 keys (V transposed)
    __shared__ unsigned int ldsP[8 * 16 * 32];    // per-wave 16x64 bf16 P tile

    const int t    = threadIdx.x;
    const int wid  = t >> 5;                      // 0..7
    const int lane = t & 31;
    const int half = lane >> 4;
    const int idx  = lane & 15;

    const int nq = SEQ / 128;
    const int qb = blockIdx.x % nq;
    const int h  = (blockIdx.x / nq) % HEADS;
    const int b  = blockIdx.x / (nq * HEADS);
    const int q0 = qb * 128;

    // stage Q block 128x64 bf16: pure b128 copies, 4 per thread
#pragma unroll
    for (int i = 0; i < 4; ++i) {
        int l = t + i * 256;                     // 0..1023
        int r = l >> 3, c8 = l & 7;
        const unsigned short* qp =
            qkv + (size_t)(b * SEQ + q0 + r) * QKV_N + h * DDIM + c8 * 8;
        *(uint4*)&ldsQ[r * 32 + c8 * 4] = *(const uint4*)qp;
    }

    const v16bf ones = ones_frag();

    // running stats in the RAW dot-product domain; scale folded into exp2
    float mrun[8];
    v8f   oacc[4], lacc;
#pragma unroll
    for (int r = 0; r < 8; ++r) mrun[r] = -3.0e38f;
#pragma unroll
    for (int d = 0; d < 4; ++d) oacc[d] = v8f{0.f,0.f,0.f,0.f,0.f,0.f,0.f,0.f};
    lacc = v8f{0.f,0.f,0.f,0.f,0.f,0.f,0.f,0.f};

    for (int kt = 0; kt < SEQ / 64; ++kt) {
        const int k0 = kt * 64;
        __syncthreads();
        // stage K tile 64x64 bf16 (row-major): 2 b128 copies per thread
#pragma unroll
        for (int i = 0; i < 2; ++i) {
            int l = t + i * 256;                  // 0..511
            int r = l >> 3, c8 = l & 7;
            const unsigned short* kp =
                qkv + (size_t)(b * SEQ + k0 + r) * QKV_N + HD + h * DDIM + c8 * 8;
            if (k0 + 64 < SEQ) __builtin_prefetch(kp + (size_t)64 * QKV_N, 0, 1);
            *(uint4*)&ldsK[r * 32 + c8 * 4] = *(const uint4*)kp;
        }
        // stage V tile 64x64 bf16 transposed [d][key]: 1 row-pair unit per thread
        {
            int kp2 = t >> 3, d8 = (t & 7) * 8;   // key-pair 0..31, d 0..56
            const unsigned short* vp =
                qkv + (size_t)(b * SEQ + k0 + kp2 * 2) * QKV_N + 2 * HD + h * DDIM + d8;
            if (k0 + 64 < SEQ) __builtin_prefetch(vp + (size_t)64 * QKV_N, 0, 1);
            const uint4 e0 = *(const uint4*)vp;
            const uint4 e1 = *(const uint4*)(vp + QKV_N);
            trans_pair_store(&ldsVt[d8 * 32 + kp2], 32, e0, e1);
        }
        __syncthreads();

        // S = Q @ K^T : 16 queries x 64 keys, 4 col-tiles x 2 k-steps (raw scores)
        v16bf qa[2];
#pragma unroll
        for (int ks = 0; ks < 2; ++ks)
            qa[ks] = load_a_frag(&ldsQ[(wid * 16 + idx) * 32 + ks * 16], half);
        v8f s[4];
#pragma unroll
        for (int ct = 0; ct < 4; ++ct) {
            s[ct] = v8f{0.f,0.f,0.f,0.f,0.f,0.f,0.f,0.f};
#pragma unroll
            for (int ks = 0; ks < 2; ++ks) {
                v16bf bb = load_b_frag(&ldsK[(ct * 16 + idx) * 32 + ks * 16], half);
                s[ct] = __builtin_amdgcn_wmma_f32_16x16x32_bf16(
                    false, qa[ks], false, bb, (short)0, s[ct], false, false);
            }
        }

        // online softmax: row max via shuffles, P stored as bf16; row sums via WMMA
        unsigned short* pw = (unsigned short*)ldsP + wid * 1024;
#pragma unroll
        for (int r = 0; r < 8; ++r) {
            float mt = fmaxf(fmaxf(s[0][r], s[1][r]), fmaxf(s[2][r], s[3][r]));
            mt = fmaxf(mt, __shfl_xor(mt, 8));
            mt = fmaxf(mt, __shfl_xor(mt, 4));
            mt = fmaxf(mt, __shfl_xor(mt, 2));
            mt = fmaxf(mt, __shfl_xor(mt, 1));
            float mnew = fmaxf(mrun[r], mt);
            float fac  = fast_exp2((mrun[r] - mnew) * EXP2_SCALE);
            mrun[r] = mnew;
#pragma unroll
            for (int ct = 0; ct < 4; ++ct) {
                float pv = fast_exp2((s[ct][r] - mnew) * EXP2_SCALE);
                pw[(half * 8 + r) * 64 + ct * 16 + idx] = f32_bf16(pv);
            }
            lacc[r] *= fac;
#pragma unroll
            for (int d = 0; d < 4; ++d) oacc[d][r] *= fac;
        }
        __syncthreads();                          // P bf16 visible in LDS

        // O += P @ V and l += P @ ones : per k-step one A-frag feeds 5 WMMAs
#pragma unroll
        for (int ks = 0; ks < 2; ++ks) {
            v16bf pa = load_a_frag(&ldsP[wid * 512 + idx * 32 + ks * 16], half);
#pragma unroll
            for (int d = 0; d < 4; ++d) {
                v16bf bb = load_b_frag(&ldsVt[(d * 16 + idx) * 32 + ks * 16], half);
                oacc[d] = __builtin_amdgcn_wmma_f32_16x16x32_bf16(
                    false, pa, false, bb, (short)0, oacc[d], false, false);
            }
            lacc = __builtin_amdgcn_wmma_f32_16x16x32_bf16(
                false, pa, false, ones, (short)0, lacc, false, false);
        }
    }

    // epilogue: O * (1/l), write bf16 [B,S,H*D]
    float linv[8];
#pragma unroll
    for (int r = 0; r < 8; ++r) linv[r] = fast_rcp(lacc[r]);
#pragma unroll
    for (int d = 0; d < 4; ++d) {
        int grow0 = q0 + wid * 16 + half * 8;
        int col   = d * 16 + idx;
        unsigned short* op = out + (size_t)(b * SEQ + grow0) * HD + h * DDIM + col;
#pragma unroll
        for (int r = 0; r < 8; ++r) op[r * HD] = f32_bf16(oacc[d][r] * linv[r]);
    }
}

extern "C" void kernel_launch(void* const* d_in, const int* in_sizes, int n_in,
                              void* d_out, int out_size, void* d_ws, size_t ws_size,
                              hipStream_t stream) {
    (void)in_sizes; (void)n_in; (void)out_size; (void)ws_size;
    const float* node = (const float*)d_in[0];   // [B,S,E]
    const float* Wqkv = (const float*)d_in[1];   // [E, 3*H*D]
    const float* Wout = (const float*)d_in[2];   // [H*D, E]
    float* out  = (float*)d_out;                 // [B,S,E]

    const int M = BATCH * SEQ;                   // 4096

    // bf16 workspace layout
    unsigned short* nodeb = (unsigned short*)d_ws;                 // M x E
    unsigned short* Wqkvb = nodeb + (size_t)M * EDIM;              // E x 3HD
    unsigned short* Woutb = Wqkvb + (size_t)EDIM * QKV_N;          // HD x E
    unsigned short* qkvb  = Woutb + (size_t)HD * EDIM;             // M x 3HD
    unsigned short* attnb = qkvb  + (size_t)M * QKV_N;             // M x HD

    // 0) one-time f32 -> bf16 conversions
    {
        int n4;
        n4 = (M * EDIM) / 4;
        cvt_f32_bf16<<<(n4 + 255) / 256, 256, 0, stream>>>(node, nodeb, n4);
        n4 = (EDIM * QKV_N) / 4;
        cvt_f32_bf16<<<(n4 + 255) / 256, 256, 0, stream>>>(Wqkv, Wqkvb, n4);
        n4 = (HD * EDIM) / 4;
        cvt_f32_bf16<<<(n4 + 255) / 256, 256, 0, stream>>>(Wout, Woutb, n4);
    }

    // 1) qkv = node @ W_qkv  (bf16 out)
    gemm_bf16_wmma<QKV_N, EDIM><<<dim3(QKV_N / 128, M / 128), 256, 0, stream>>>(
        nodeb, Wqkvb, qkvb);
    // 2) flash attention per (b, h, 128-query block) (bf16 out)
    attn_fwd<<<BATCH * HEADS * (SEQ / 128), 256, 0, stream>>>(qkvb, attnb);
    // 3) out = attn @ W_out  (f32 out)
    gemm_bf16_wmma<HD, EDIM><<<dim3(HD / 128, M / 128), 256, 0, stream>>>(
        attnb, Woutb, out);
}
